// MambaBlock_3607772529040
// MI455X (gfx1250) — matrix operand order repaired and checked by reading, not compiled, not run on hardware
//
#include <hip/hip_runtime.h>
#include <math.h>

// ---------------------------------------------------------------------------
// Mamba block for MI455X (gfx1250, wave32, WMMA + async LDS staging).
//   B=2, T=1024, DM=1024, D=2048, DTR=128, N=16, K=4
// GEMMs: f16 operands (global) -> double-buffered LDS (async A) ->
//        v_wmma_f32_16x16x32_f16, f32 accumulate.
// ---------------------------------------------------------------------------

typedef __attribute__((ext_vector_type(16))) _Float16 v16h;
typedef __attribute__((ext_vector_type(8)))  float    v8f;

union F16Frag { v16h v; unsigned int u[8]; };
union F32Acc  { v8f  v; float f[8]; };
union H4      { _Float16 h[4]; uint2 u2; };

#define BM 128
#define BN 64
#define BK 32
#define LDP (BK + 8)           // 40 halfs = 80B row stride (multiple of 16B,
                               // 20-bank stride -> conflict-free b128 gathers)
#define SBUF (BM * LDP + BN * LDP)   // halfs per LDS stage (A then B)

// epilogue modes
#define EPI_NONE     0
#define EPI_SOFTPLUS 1
#define EPI_RESID    2

// ---------------------------------------------------------------------------
// C[M x Nn] = A16[M x Kdim](lda) * B16[Kdim x Nn]  (+bias, epilogue)
// Grid: (ceil(Nn/BN), M/BM). Block: 256 threads = 8 waves (4 along M, 2 N).
// Double-buffered LDS: tile k+1 is staged (async A via ASYNCcnt; B via one
// uint4 load + 8-half scatter) while tile k is consumed by WMMA.
// ---------------------------------------------------------------------------
__global__ __launch_bounds__(256)
void gemm_f16wmma(const _Float16* __restrict__ A, int lda,
                  const _Float16* __restrict__ Bw, int Nn, int Kdim,
                  const float* __restrict__ bias,
                  const float* __restrict__ resid,
                  float* __restrict__ C,
                  _Float16* __restrict__ C16,
                  int mode)
{
    // single LDS object so async byte offsets are well-defined (base = 0)
    __shared__ __align__(16) _Float16 smem[2 * SBUF];

    const int tid  = threadIdx.x;
    const int lane = tid & 31;
    const int wave = tid >> 5;
    const int wm   = wave & 3;
    const int wn   = wave >> 2;
    const int bM   = blockIdx.y * BM;
    const int bN   = blockIdx.x * BN;
    const int grp  = lane >> 4;
    const int lm   = lane & 15;

    // ---- A tile: 128 rows x 64B; 512 16B chunks -> 2 per thread ----
    const int ar   = tid >> 2;            // 0..63
    const int aseg = tid & 3;             // 16B segment in row
    const _Float16* pA0 = A + (size_t)(bM + ar)      * lda + aseg * 8;
    const _Float16* pA1 = A + (size_t)(bM + 64 + ar) * lda + aseg * 8;
    const unsigned ldsA0 = (unsigned)(ar        * (LDP * 2) + aseg * 16);
    const unsigned ldsA1 = (unsigned)((64 + ar) * (LDP * 2) + aseg * 16);

    // ---- B tile: 32 rows x 128B; 256 16B chunks -> 1 per thread ----
    // Out-of-range N columns are clamped (harmless garbage in LDS; the
    // epilogue store is guarded), avoiding any divergent load path.
    const int bk   = tid >> 3;            // k row 0..31
    const int bn0  = (tid & 7) * 8;       // first of 8 columns
    const int bnc  = (bN + bn0 < Nn) ? (bN + bn0) : (Nn - 8);
    const _Float16* pB = Bw + (size_t)bk * Nn + bnc;

    // async A issue into stage s
    auto issueA = [&](int s, const _Float16* a0, const _Float16* a1) {
        const unsigned so = (unsigned)(s * (SBUF * 2));
        asm volatile("global_load_async_to_lds_b128 %0, %1, off"
                     :: "v"(ldsA0 + so), "v"(a0) : "memory");
        asm volatile("global_load_async_to_lds_b128 %0, %1, off"
                     :: "v"(ldsA1 + so), "v"(a1) : "memory");
    };
    // B scatter-transpose into stage s
    auto scatterB = [&](int s, const uint4& bxu) {
        _Float16* lBn = smem + s * SBUF + BM * LDP;
        const _Float16* bh = (const _Float16*)&bxu;
#pragma unroll
        for (int j = 0; j < 8; ++j)
            lBn[(bn0 + j) * LDP + bk] = bh[j];
    };

    F32Acc acc[2][2];
#pragma unroll
    for (int tm = 0; tm < 2; ++tm)
#pragma unroll
        for (int tn = 0; tn < 2; ++tn)
#pragma unroll
            for (int v = 0; v < 8; ++v) acc[tm][tn].f[v] = 0.0f;

    // ---- prologue: stage tile 0 into buffer 0 ----
    issueA(0, pA0, pA1);
    {
        const uint4 b0 = *(const uint4*)pB;
        scatterB(0, b0);
    }
    asm volatile("s_wait_asynccnt 0" ::: "memory");
    __syncthreads();

    const int nt = Kdim / BK;
    for (int it = 0; it < nt; ++it) {
        const int cur = it & 1;
        const int nxt = cur ^ 1;
        const bool have_next = (it + 1 < nt);

        // ---- issue next tile's loads first (overlap with WMMA below) ----
        uint4 bxu = make_uint4(0u, 0u, 0u, 0u);
        if (have_next) {
            pA0 += BK; pA1 += BK; pB += (size_t)BK * Nn;
            issueA(nxt, pA0, pA1);
            bxu = *(const uint4*)pB;                         // LOADcnt in flight
            __builtin_prefetch(pB + (size_t)BK * Nn, 0, 3);  // speculative (SE)
        }

        // ---- fragments + WMMA on current buffer ----
        const _Float16* lAc = smem + cur * SBUF;
        const _Float16* lBc = lAc + BM * LDP;
        F16Frag af[2], bf[2];
#pragma unroll
        for (int tm = 0; tm < 2; ++tm) {
            const int m = wm * 32 + tm * 16 + lm;
#pragma unroll
            for (int v = 0; v < 8; ++v) {
                const int kk = ((v < 4) ? 0 : 16) + grp * 8 + (v & 3) * 2;
                af[tm].u[v] = *(const unsigned int*)&lAc[m * LDP + kk];
            }
        }
#pragma unroll
        for (int tn = 0; tn < 2; ++tn) {
            const int n = wn * 32 + tn * 16 + lm;
#pragma unroll
            for (int v = 0; v < 8; ++v) {
                const int kk = ((v < 4) ? 0 : 16) + grp * 8 + (v & 3) * 2;
                bf[tn].u[v] = *(const unsigned int*)&lBc[n * LDP + kk];
            }
        }
#pragma unroll
        for (int tm = 0; tm < 2; ++tm)
#pragma unroll
            for (int tn = 0; tn < 2; ++tn)
                acc[tm][tn].v = __builtin_amdgcn_wmma_f32_16x16x32_f16(
                    false, af[tm].v, false, bf[tn].v,
                    (short)0, acc[tm][tn].v, false, false);

        // ---- finish staging next buffer, then rendezvous ----
        if (have_next) {
            scatterB(nxt, bxu);                       // waits LOADcnt here
            asm volatile("s_wait_asynccnt 0" ::: "memory");
        }
        __syncthreads();
    }

    // ---- epilogue: lane l -> n = l&15 ; VGPR v -> m = 8*(l>>4)+v ----
#pragma unroll
    for (int tm = 0; tm < 2; ++tm) {
#pragma unroll
        for (int tn = 0; tn < 2; ++tn) {
            const int n = bN + wn * 32 + tn * 16 + lm;
            if (n < Nn) {
                const float bv = bias ? bias[n] : 0.0f;
#pragma unroll
                for (int v = 0; v < 8; ++v) {
                    const int m = bM + wm * 32 + tm * 16 + grp * 8 + v;
                    float val = acc[tm][tn].f[v] + bv;
                    if (mode == EPI_SOFTPLUS) {
                        val = (val > 20.0f) ? val : log1pf(expf(val));
                    } else if (mode == EPI_RESID) {
                        val += resid[(size_t)m * Nn + n];
                    }
                    C[(size_t)m * Nn + n] = val;
                    if (C16) C16[(size_t)m * Nn + n] = (_Float16)val;
                }
            }
        }
    }
}

// ---------------------------------------------------------------------------
// f32 -> f16 bulk convert (weights). count must be a multiple of 1024.
// ---------------------------------------------------------------------------
__global__ __launch_bounds__(256)
void f32_to_f16_kernel(const float* __restrict__ s, _Float16* __restrict__ d)
{
    const size_t i = (size_t)(blockIdx.x * 256 + threadIdx.x) * 4;
    const float4 f = *(const float4*)(s + i);
    H4 h;
    h.h[0] = (_Float16)f.x; h.h[1] = (_Float16)f.y;
    h.h[2] = (_Float16)f.z; h.h[3] = (_Float16)f.w;
    *(uint2*)(d + i) = h.u2;
}

// ---------------------------------------------------------------------------
// RMSNorm over last dim (1024): one block per row, f16 output (GEMM A operand).
// ---------------------------------------------------------------------------
__global__ __launch_bounds__(256)
void rmsnorm_kernel(const float* __restrict__ x, const float* __restrict__ w,
                    _Float16* __restrict__ out16)
{
    const int row = blockIdx.x;
    const int tid = threadIdx.x;
    const float* xr = x + (size_t)row * 1024;
    float v[4];
    float s = 0.0f;
#pragma unroll
    for (int i = 0; i < 4; ++i) {
        v[i] = xr[tid + i * 256];
        s += v[i] * v[i];
    }
    __shared__ float red[256];
    red[tid] = s;
    __syncthreads();
    for (int off = 128; off >= 1; off >>= 1) {
        if (tid < off) red[tid] += red[tid + off];
        __syncthreads();
    }
    const float rms = rsqrtf(red[0] * (1.0f / 1024.0f) + 1e-6f);
    _Float16* orow = out16 + (size_t)row * 1024;
#pragma unroll
    for (int i = 0; i < 4; ++i)
        orow[tid + i * 256] = (_Float16)(v[i] * rms * w[tid + i * 256]);
}

// ---------------------------------------------------------------------------
// Depthwise causal conv (K=4) + bias + SiLU. Dual output: f32 (scan) + f16 (GEMM).
// ---------------------------------------------------------------------------
__global__ __launch_bounds__(256)
void conv_silu_kernel(const float* __restrict__ proj,
                      const float* __restrict__ conv_w,
                      const float* __restrict__ conv_b,
                      float* __restrict__ x1cF,
                      _Float16* __restrict__ x1cH)
{
    const int idx = blockIdx.x * 256 + threadIdx.x;   // 0 .. 2*1024*2048-1
    const int d = idx & 2047;
    const int t = (idx >> 11) & 1023;
    const int b = idx >> 21;
    float acc = conv_b[d];
#pragma unroll
    for (int j = 0; j < 4; ++j) {
        const int tq = t - 3 + j;
        if (tq >= 0)
            acc += proj[(size_t)((b << 10) + tq) * 4096 + d] * conv_w[d * 4 + j];
    }
    const float s = acc / (1.0f + expf(-acc));        // SiLU
    const size_t o = (size_t)((b << 10) + t) * 2048 + d;
    x1cF[o] = s;
    x1cH[o] = (_Float16)s;
}

// ---------------------------------------------------------------------------
// Selective scan. One thread per (b,d,n): 65536 threads, 2048 wave32 waves.
//   h = exp(delta*A)*h + delta*B*x ; y = sum_n h*C ; out = silu(g)*y  (f16)
// ---------------------------------------------------------------------------
__global__ __launch_bounds__(256)
void scan_kernel(const float* __restrict__ proj,     // g at [..., 2048+d], stride 4096
                 const float* __restrict__ x1c,      // stride 2048
                 const float* __restrict__ delta,    // stride 2048
                 const float* __restrict__ dbc,      // stride 160; B at +128, C at +144
                 const float* __restrict__ A_log,    // (2048,16)
                 _Float16* __restrict__ yg16)        // stride 2048
{
    const int gid = blockIdx.x * 256 + threadIdx.x;  // 0..65535
    const int n  = gid & 15;
    const int bd = gid >> 4;
    const int d  = bd & 2047;
    const int b  = bd >> 11;
    const float a = -expf(A_log[d * 16 + n]);
    float h = 0.0f;
    for (int t = 0; t < 1024; ++t) {
        const size_t r = (size_t)((b << 10) + t);
        const float dlt = delta[r * 2048 + d];
        const float xv  = x1c [r * 2048 + d];
        const float bm  = dbc[r * 160 + 128 + n];
        const float cm  = dbc[r * 160 + 144 + n];
        h = expf(dlt * a) * h + (dlt * bm) * xv;
        float y = h * cm;
        y += __shfl_xor(y, 1, 32);
        y += __shfl_xor(y, 2, 32);
        y += __shfl_xor(y, 4, 32);
        y += __shfl_xor(y, 8, 32);
        if (n == 0) {
            const float g = proj[r * 4096 + 2048 + d];
            yg16[r * 2048 + d] = (_Float16)(y * (g / (1.0f + expf(-g))));
        }
    }
}

// ---------------------------------------------------------------------------
extern "C" void kernel_launch(void* const* d_in, const int* in_sizes, int n_in,
                              void* d_out, int out_size, void* d_ws, size_t ws_size,
                              hipStream_t stream)
{
    const float* x      = (const float*)d_in[0];   // (2,1024,1024)
    const float* rms_w  = (const float*)d_in[1];   // (1024)
    const float* in_W   = (const float*)d_in[2];   // (1024,4096)
    const float* in_b   = (const float*)d_in[3];   // (4096)
    const float* conv_w = (const float*)d_in[4];   // (2048,4)
    const float* conv_b = (const float*)d_in[5];   // (2048)
    const float* A_log  = (const float*)d_in[6];   // (2048,16)
    const float* dbc_W  = (const float*)d_in[7];   // (2048,160)
    const float* dbc_b  = (const float*)d_in[8];   // (160)
    const float* dup_W  = (const float*)d_in[9];   // (128,2048)
    const float* dup_b  = (const float*)d_in[10];  // (2048)
    const float* out_W  = (const float*)d_in[11];  // (2048,1024)
    const float* out_b  = (const float*)d_in[12];  // (1024)
    float* out = (float*)d_out;                    // (2,1024,1024) f32
    const size_t M = 2048;                         // 2*1024 rows

    // ---- workspace carve-up (256B aligned regions) ----
    char* wp = (char*)d_ws;
    auto carve = [&](size_t bytes) { void* r = (void*)wp; wp += (bytes + 255) & ~(size_t)255; return r; };
    float*    proj   = (float*)   carve(M * 4096 * 4);
    float*    x1c    = (float*)   carve(M * 2048 * 4);
    float*    dbc    = (float*)   carve(M * 160 * 4);
    float*    delta  = (float*)   carve(M * 2048 * 4);
    _Float16* hn16   = (_Float16*)carve(M * 1024 * 2);
    _Float16* x1c16  = (_Float16*)carve(M * 2048 * 2);
    _Float16* dbc16  = (_Float16*)carve(M * 160 * 2);
    _Float16* yg16   = (_Float16*)carve(M * 2048 * 2);
    _Float16* inW16  = (_Float16*)carve((size_t)1024 * 4096 * 2);
    _Float16* dbcW16 = (_Float16*)carve((size_t)2048 * 160 * 2);
    _Float16* dupW16 = (_Float16*)carve((size_t)128 * 2048 * 2);
    _Float16* outW16 = (_Float16*)carve((size_t)2048 * 1024 * 2);

    // 0) weights -> f16 (once per launch; deterministic)
    f32_to_f16_kernel<<<(1024 * 4096) / 1024, 256, 0, stream>>>(in_W,  inW16);
    f32_to_f16_kernel<<<(2048 * 160)  / 1024, 256, 0, stream>>>(dbc_W, dbcW16);
    f32_to_f16_kernel<<<(128 * 2048)  / 1024, 256, 0, stream>>>(dup_W, dupW16);
    f32_to_f16_kernel<<<(2048 * 1024) / 1024, 256, 0, stream>>>(out_W, outW16);

    // 1) RMSNorm -> f16
    rmsnorm_kernel<<<2048, 256, 0, stream>>>(x, rms_w, hn16);
    // 2) in-proj: proj = hn @ in_W + in_b     [2048 x 1024 x 4096]
    gemm_f16wmma<<<dim3(4096 / BN, 2048 / BM), 256, 0, stream>>>(
        hn16, 1024, inW16, 4096, 1024, in_b, nullptr, proj, nullptr, EPI_NONE);
    // 3) depthwise causal conv + SiLU -> x1c (f32 + f16)
    conv_silu_kernel<<<(2 * 1024 * 2048) / 256, 256, 0, stream>>>(
        proj, conv_w, conv_b, x1c, x1c16);
    // 4) dbc = x1c @ dbc_W + dbc_b            [2048 x 2048 x 160] (guarded N)
    gemm_f16wmma<<<dim3((160 + BN - 1) / BN, 2048 / BM), 256, 0, stream>>>(
        x1c16, 2048, dbcW16, 160, 2048, dbc_b, nullptr, dbc, dbc16, EPI_NONE);
    // 5) delta = softplus(dbc[:, :128] @ dup_W + dup_b)  [2048 x 128 x 2048]
    gemm_f16wmma<<<dim3(2048 / BN, 2048 / BM), 256, 0, stream>>>(
        dbc16, 160, dupW16, 2048, 128, dup_b, nullptr, delta, nullptr, EPI_SOFTPLUS);
    // 6) selective scan + gating -> yg (f16)
    scan_kernel<<<256, 256, 0, stream>>>(proj, x1c, delta, dbc, A_log, yg16);
    // 7) out = yg @ out_W + out_b + residual(x)  [2048 x 2048 x 1024]
    gemm_f16wmma<<<dim3(1024 / BN, 2048 / BM), 256, 0, stream>>>(
        yg16, 2048, outW16, 1024, 2048, out_b, x, out, nullptr, EPI_RESID);
}
